// Net_PDE_35124242547210
// MI455X (gfx1250) — compile-verified, matrix operand order checked by reading, not run on hardware
//
#include <hip/hip_runtime.h>
#include <hip/hip_bf16.h>

// ---------------------------------------------------------------------------
// PINN residual for a 6-layer tanh MLP, width 256, via forward jet propagation.
// Jet channels: z, z_x, z_y, z_t, z_xx, z_yy  -> 6 rows per point share each W.
// Interior layers (w1..w5, 256x256) run as bf16 WMMA GEMMs on the 96x256 jet,
// with the weight (B) fragment held in registers and reused across all 6
// channel tiles (6 resident f32 accumulators per wave).
// ---------------------------------------------------------------------------

typedef __bf16  v16bf __attribute__((ext_vector_type(16)));
typedef float   v8f   __attribute__((ext_vector_type(8)));

#define PTS      16          // points per workgroup
#define NCH      6           // jet channels
#define MROWS    (NCH*PTS)   // 96 rows in the jet matrix
#define WID      256
#define ZSTR     264         // bf16 row stride (pad -> conflict-free b128 frags)
#define USTR     260         // f32  row stride (pad -> conflict-free b32)
#define Z_BYTES  (MROWS*ZSTR*2)          // 50688
#define U_BYTES  (MROWS*USTR*4)          // 99840
#define MU_C     1.0f
#define K_COND_C 0.5f
#define PI_C     3.14159265358979323846f

union FragBF { v16bf v; uint4 q[2]; };

static __device__ __forceinline__ unsigned short f2bf(float f) {
    unsigned int u = __float_as_uint(f);
    u += 0x7FFFu + ((u >> 16) & 1u);           // round-to-nearest-even
    return (unsigned short)(u >> 16);
}
static __device__ __forceinline__ float bf2f(unsigned short h) {
    return __uint_as_float(((unsigned int)h) << 16);
}

// Fast tanh on the hardware transcendental path: v_exp_f32 + v_rcp_f32.
// Clamped so exp never overflows; accuracy ~1 ulp of bf16 pipeline needs.
static __device__ __forceinline__ float fast_tanh(float x) {
    float cx = fminf(fmaxf(x, -15.0f), 15.0f);
    float e  = __expf(2.0f * cx);
    return 1.0f - 2.0f * __builtin_amdgcn_rcpf(e + 1.0f);
}

// ---- prepack: w1..w5 (f32 row-major [k][n]) -> bf16 transposed [l][n][k] ----
__global__ void pinn_prepack(const float* __restrict__ w1, const float* __restrict__ w2,
                             const float* __restrict__ w3, const float* __restrict__ w4,
                             const float* __restrict__ w5, unsigned short* __restrict__ wp) {
    int tid = blockIdx.x * blockDim.x + threadIdx.x;
    if (tid >= 5 * 65536) return;
    int l = tid >> 16;
    int e = tid & 65535;
    int n = e >> 8;
    int k = e & 255;
    const float* w = (l == 0) ? w1 : (l == 1) ? w2 : (l == 2) ? w3 : (l == 3) ? w4 : w5;
    wp[(size_t)l * 65536 + n * 256 + k] = f2bf(w[k * 256 + n]);
}

// ---- elementwise tanh-chain over the jet (writes Z bf16) -------------------
static __device__ __forceinline__ void tanh_chain_store(
    unsigned short* Z, int p, int n,
    float u, float ux, float uy, float ut, float uxx, float uyy) {
    float a  = fast_tanh(u);
    float s  = 1.0f - a * a;
    float g  = -2.0f * a * s;
    float zx = s * ux, zy = s * uy, zt = s * ut;
    float zxx = s * uxx + g * ux * ux;
    float zyy = s * uyy + g * uy * uy;
    Z[(0 * PTS + p) * ZSTR + n] = f2bf(a);
    Z[(1 * PTS + p) * ZSTR + n] = f2bf(zx);
    Z[(2 * PTS + p) * ZSTR + n] = f2bf(zy);
    Z[(3 * PTS + p) * ZSTR + n] = f2bf(zt);
    Z[(4 * PTS + p) * ZSTR + n] = f2bf(zxx);
    Z[(5 * PTS + p) * ZSTR + n] = f2bf(zyy);
}

__global__ void __launch_bounds__(256, 2) pinn_residual(
    const float* __restrict__ xyt,
    const float* __restrict__ w0, const float* __restrict__ b0,
    const float* __restrict__ b1, const float* __restrict__ b2,
    const float* __restrict__ b3, const float* __restrict__ b4,
    const float* __restrict__ b5,
    const float* __restrict__ w6, const float* __restrict__ b6,
    const unsigned short* __restrict__ wpack,
    float* __restrict__ out) {

    extern __shared__ char smem[];
    unsigned short* Z = (unsigned short*)smem;          // 96 x 264 bf16
    float*          U = (float*)(smem + Z_BYTES);       // 96 x 260 f32

    const int tid  = threadIdx.x;
    const int wave = tid >> 5;
    const int lane = tid & 31;
    const int r    = lane & 15;
    const int hi   = lane >> 4;          // 0: lanes 0-15, 1: lanes 16-31
    const int pblk = blockIdx.x * PTS;

    // Elementwise-phase mapping: lane stride == row stride -> conflict-free LDS.
    const int ep = tid & 15;             // point
    const int en = tid >> 4;             // feature-phase 0..15

    // ---------------- layer 0: (x,y,t) -> 256, analytic jet ----------------
    {
        int pg = pblk + ep;
        float x = xyt[pg * 3 + 0];
        float y = xyt[pg * 3 + 1];
        float t = xyt[pg * 3 + 2];
        #pragma unroll 4
        for (int j = 0; j < 16; ++j) {
            int n = en + j * 16;
            float c0 = w0[0 * WID + n];
            float c1 = w0[1 * WID + n];
            float c2 = w0[2 * WID + n];
            float u  = x * c0 + y * c1 + t * c2 + b0[n];
            // u_x=c0, u_y=c1, u_t=c2, u_xx=u_yy=0
            tanh_chain_store(Z, ep, n, u, c0, c1, c2, 0.0f, 0.0f);
        }
    }

    // ---------------- layers 1..5: bf16 WMMA GEMM + tanh chain -------------
    const float* bs[5] = { b1, b2, b3, b4, b5 };
    for (int l = 0; l < 5; ++l) {
        __syncthreads();                                   // Z ready
        const unsigned short* Wl = wpack + (size_t)l * 65536;
        __builtin_prefetch(Wl + wave * 8192, 0, 0);

        // U = Z @ W : each wave owns one 16-col N-tile at a time, keeps all
        // six channel accumulators resident, and reuses the B fragment 6x.
        for (int ng = wave; ng < 16; ng += 8) {
            const int nb = ng * 16;
            v8f acc[NCH];
            #pragma unroll
            for (int m = 0; m < NCH; ++m)
                acc[m] = (v8f){0.f, 0.f, 0.f, 0.f, 0.f, 0.f, 0.f, 0.f};

            const unsigned short* bp0 = Wl + (nb + r) * 256 + hi * 8;
            const unsigned short* ap0 = &Z[r * ZSTR + hi * 8];
            #pragma unroll
            for (int kk = 0; kk < 8; ++kk) {
                const int kb = kk * 32;
                FragBF b;
                b.q[0] = *(const uint4*)(bp0 + kb);
                b.q[1] = *(const uint4*)(bp0 + kb + 16);
                #pragma unroll
                for (int m = 0; m < NCH; ++m) {
                    FragBF a;
                    const unsigned short* ap = ap0 + m * PTS * ZSTR + kb;
                    a.q[0] = *(const uint4*)(ap);
                    a.q[1] = *(const uint4*)(ap + 16);
                    acc[m] = __builtin_amdgcn_wmma_f32_16x16x32_bf16(
                                 false, a.v, false, b.v, (short)0, acc[m],
                                 false, false);
                }
            }
            // C layout: lane r(+hi*16) holds column nb+r, rows m*16 + hi*8 + v
            #pragma unroll
            for (int m = 0; m < NCH; ++m) {
                int rowBase = m * PTS + hi * 8;
                #pragma unroll
                for (int v = 0; v < 8; ++v)
                    U[(rowBase + v) * USTR + nb + r] = acc[m][v];
            }
        }
        __syncthreads();                                   // U ready

        const float* bl = bs[l];
        #pragma unroll 4
        for (int j = 0; j < 16; ++j) {
            int n = en + j * 16;
            float u   = U[(0 * PTS + ep) * USTR + n] + bl[n];
            float ux  = U[(1 * PTS + ep) * USTR + n];
            float uy  = U[(2 * PTS + ep) * USTR + n];
            float ut  = U[(3 * PTS + ep) * USTR + n];
            float uxx = U[(4 * PTS + ep) * USTR + n];
            float uyy = U[(5 * PTS + ep) * USTR + n];
            tanh_chain_store(Z, ep, n, u, ux, uy, ut, uxx, uyy);
        }
    }

    // ---------------- layer 6: 256 -> 1 dot products + PDE residual --------
    __syncthreads();                                       // Z final, U free
    {
        int p   = tid & 15;
        int seg = tid >> 4;                // 16 features per thread
        float s0 = 0.f, s1 = 0.f, s2 = 0.f, s3 = 0.f, s4 = 0.f, s5 = 0.f;
        #pragma unroll
        for (int i = 0; i < 16; ++i) {
            int n = seg * 16 + i;
            float wv = w6[n];
            s0 += bf2f(Z[(0 * PTS + p) * ZSTR + n]) * wv;
            s1 += bf2f(Z[(1 * PTS + p) * ZSTR + n]) * wv;
            s2 += bf2f(Z[(2 * PTS + p) * ZSTR + n]) * wv;
            s3 += bf2f(Z[(3 * PTS + p) * ZSTR + n]) * wv;
            s4 += bf2f(Z[(4 * PTS + p) * ZSTR + n]) * wv;
            s5 += bf2f(Z[(5 * PTS + p) * ZSTR + n]) * wv;
        }
        float* red = U;                    // reuse U as 96x16 partials
        red[(0 * PTS + p) * 16 + seg] = s0;
        red[(1 * PTS + p) * 16 + seg] = s1;
        red[(2 * PTS + p) * 16 + seg] = s2;
        red[(3 * PTS + p) * 16 + seg] = s3;
        red[(4 * PTS + p) * 16 + seg] = s4;
        red[(5 * PTS + p) * 16 + seg] = s5;
    }
    __syncthreads();
    if (tid < PTS) {
        int p = tid;
        float acc[NCH];
        #pragma unroll
        for (int c = 0; c < NCH; ++c) {
            float s = 0.f;
            #pragma unroll
            for (int seg = 0; seg < 16; ++seg)
                s += U[(c * PTS + p) * 16 + seg];
            acc[c] = s;
        }
        float h   = acc[0] + b6[0];
        float hx  = acc[1], hy = acc[2], ht = acc[3];
        float hxx = acc[4], hyy = acc[5];
        int pg = pblk + p;
        float x = xyt[pg * 3 + 0];
        float y = xyt[pg * 3 + 1];
        float t = xyt[pg * 3 + 2];
        float f = sinf(PI_C * x) * sinf(PI_C * y) * expf(-t);
        out[pg] = MU_C * ht
                - K_COND_C * (h * (hxx + hyy) + hx * hx + hy * hy)
                - f;
    }
}

extern "C" void kernel_launch(void* const* d_in, const int* in_sizes, int n_in,
                              void* d_out, int out_size, void* d_ws, size_t ws_size,
                              hipStream_t stream) {
    (void)in_sizes; (void)n_in; (void)out_size; (void)ws_size;
    const float* xyt = (const float*)d_in[0];
    const float* W[7];
    const float* B[7];
    for (int i = 0; i < 7; ++i) {
        W[i] = (const float*)d_in[1 + 2 * i];
        B[i] = (const float*)d_in[2 + 2 * i];
    }
    unsigned short* wpack = (unsigned short*)d_ws;   // 5 * 65536 bf16 = 640 KB

    pinn_prepack<<<(5 * 65536 + 255) / 256, 256, 0, stream>>>(
        W[1], W[2], W[3], W[4], W[5], wpack);

    const size_t smem = (size_t)Z_BYTES + (size_t)U_BYTES;   // ~147 KB
    const int nblocks = 131072 / PTS;                        // 8192
    pinn_residual<<<nblocks, 256, smem, stream>>>(
        xyt, W[0], B[0], B[1], B[2], B[3], B[4], B[5],
        W[6], B[6], wpack, (float*)d_out);
}
